// OracleAD2D_46505905881368
// MI455X (gfx1250) — matrix operand order repaired and checked by reading, not compiled, not run on hardware
//
#include <hip/hip_runtime.h>
#include <hip/hip_bf16.h>

#define NVARS 32
#define LSEQ  128
#define DDIM  128
#define FOURD 512
#define BATCH 64
#define HEADS 4
#define DH    32
#define WS    136   // padded LDS row stride in halves for D=128 rows (conflict-free b128)
#define ACCS  132   // padded f32 acc row stride

typedef __attribute__((ext_vector_type(16))) _Float16 v16h;
typedef __attribute__((ext_vector_type(8)))  _Float16 v8h;
typedef __attribute__((ext_vector_type(8)))  float    v8f;

__device__ __forceinline__ v8f wmma16(v16h a, v16h b, v8f c) {
  // D = A(16x32 f16) * B(32x16 f16) + C(16x16 f32)
  return __builtin_amdgcn_wmma_f32_16x16x32_f16(false, a, false, b, (short)0, c, false, false);
}

// A fragment (16x32 f16): lane (m=lane&15, hi=lane>>4)
// halves[0..7]  = row[kbase + hi*16 .. +8]
// halves[8..15] = row[kbase + hi*16 + 8 .. +8]
__device__ __forceinline__ v16h load_a_frag(const _Float16* row, int kbase, int hi) {
  v8h a0 = *(const v8h*)(row + kbase + hi * 16);
  v8h a1 = *(const v8h*)(row + kbase + hi * 16 + 8);
  v16h r;
#pragma unroll
  for (int i = 0; i < 8; ++i) { r[i] = a0[i]; r[i + 8] = a1[i]; }
  return r;
}

// B fragment (32x16 f16): lane holds column n=lane&15, K = hi*16 .. +16 contiguous
__device__ __forceinline__ v16h load_b_frag(const _Float16* p) {
  v8h b0 = *(const v8h*)p;
  v8h b1 = *(const v8h*)(p + 8);
  v16h r;
#pragma unroll
  for (int i = 0; i < 8; ++i) { r[i] = b0[i]; r[i + 8] = b1[i]; }
  return r;
}

// B fragment from f32 global memory with on-the-fly convert
__device__ __forceinline__ v16h load_b_frag_f32(const float* p) {
  v16h r;
#pragma unroll
  for (int i = 0; i < 16; i += 4) {
    float4 f = *(const float4*)(p + i);
    r[i] = (_Float16)f.x; r[i + 1] = (_Float16)f.y;
    r[i + 2] = (_Float16)f.z; r[i + 3] = (_Float16)f.w;
  }
  return r;
}

__device__ __forceinline__ float sigmoidf_(float x) { return 1.f / (1.f + __expf(-x)); }

// reduce across the 16-lane half (xor masks < 16 keep halves independent on wave32)
__device__ __forceinline__ float rowred_sum(float v) {
  v += __shfl_xor(v, 1, 32); v += __shfl_xor(v, 2, 32);
  v += __shfl_xor(v, 4, 32); v += __shfl_xor(v, 8, 32);
  return v;
}
__device__ __forceinline__ float rowred_max(float v) {
  v = fmaxf(v, __shfl_xor(v, 1, 32)); v = fmaxf(v, __shfl_xor(v, 2, 32));
  v = fmaxf(v, __shfl_xor(v, 4, 32)); v = fmaxf(v, __shfl_xor(v, 8, 32));
  return v;
}

// ---------------------------------------------------------------------------
// Encoder: per-(variable, batch-quarter) workgroup. 127 LSTM steps with LDS-
// resident Whh (f16), WMMA recurrence, fused online-softmax temporal pooling.
// Writes C (B,N,D) as f16 into workspace.
// ---------------------------------------------------------------------------
__global__ void enc_kernel(const float* __restrict__ X,
                           const float* __restrict__ Wih,
                           const float* __restrict__ Whh,
                           const float* __restrict__ bih,
                           const float* __restrict__ bhh,
                           const float* __restrict__ pool_w,
                           const float* __restrict__ pool_b,
                           _Float16* __restrict__ Cf16) {
  extern __shared__ char smem[];
  _Float16* Wlds = (_Float16*)smem;            // FOURD*WS halves (f16 weights)
  _Float16* Hlds = Wlds + FOURD * WS;          // 16*WS halves (h state)
  float* Acc = (float*)(Hlds + 16 * WS);       // 16*ACCS (pooled accumulator)
  float* Xs  = Acc + 16 * ACCS;                // 16
  float* Slds = Xs + 16;                       // 16 (score partials)
  float* Mst = Slds + 16;                      // 16 (running max)
  float* Sst = Mst + 16;                       // 16 (running sum)
  float* Scl = Sst + 16;                       // 16
  float* Wgt = Scl + 16;                       // 16

  const int tid = threadIdx.x;
  const int wave = tid >> 5, lane = tid & 31, m = lane & 15, hi = lane >> 4;
  const int n = blockIdx.x >> 2, bq = blockIdx.x & 3;
  const int b0 = bq * 16;

  // stage Whh[n] f32 -> f16 LDS (padded rows)
  const float* Wg = Whh + (size_t)n * FOURD * DDIM;
  for (int idx = tid; idx < FOURD * DDIM / 4; idx += 256) {
    float4 f = ((const float4*)Wg)[idx];
    int e = idx * 4, g = e >> 7, d = e & 127;
    _Float16* p = Wlds + g * WS + d;
    p[0] = (_Float16)f.x; p[1] = (_Float16)f.y; p[2] = (_Float16)f.z; p[3] = (_Float16)f.w;
  }
  for (int idx = tid; idx < 16 * WS; idx += 256) Hlds[idx] = (_Float16)0.f;
  for (int idx = tid; idx < 16 * ACCS; idx += 256) Acc[idx] = 0.f;
  if (tid < 16) { Slds[tid] = 0.f; Mst[tid] = -3.0e38f; Sst[tid] = 0.f; }

  // this wave owns hidden columns [wave*16, wave*16+16); per-lane column:
  const int dcol = wave * 16 + m;
  float wih_g[4], bias_g[4];
#pragma unroll
  for (int G = 0; G < 4; ++G) {
    int g = G * DDIM + dcol;
    wih_g[G]  = Wih[n * FOURD + g];
    bias_g[G] = bih[n * FOURD + g] + bhh[n * FOURD + g];
  }
  const float pw = pool_w[n * DDIM + dcol];
  const float pb = pool_b[n];

  float creg[8];
#pragma unroll
  for (int r = 0; r < 8; ++r) creg[r] = 0.f;

  __syncthreads();

  for (int t = 0; t < LSEQ - 1; ++t) {
    if (tid < 16) Xs[tid] = X[(size_t)(b0 + tid) * LSEQ * NVARS + t * NVARS + n];
    __syncthreads();

    float xr[8];
#pragma unroll
    for (int r = 0; r < 8; ++r) xr[r] = Xs[r + 8 * hi];

    v16h afr[4];
#pragma unroll
    for (int kc = 0; kc < 4; ++kc) afr[kc] = load_a_frag(Hlds + m * WS, kc * 32, hi);

    v8f acc[4];
#pragma unroll
    for (int G = 0; G < 4; ++G) {
#pragma unroll
      for (int r = 0; r < 8; ++r) acc[G][r] = xr[r] * wih_g[G] + bias_g[G];
    }
    // kc outer / G inner: consecutive WMMAs hit different accumulators, so the
    // f16 WMMA->WMMA result-reuse hazard slots are filled with independent WMMAs
#pragma unroll
    for (int kc = 0; kc < 4; ++kc) {
#pragma unroll
      for (int G = 0; G < 4; ++G) {
        const _Float16* brow = Wlds + (G * DDIM + dcol) * WS;
        v16h bf = load_b_frag(brow + kc * 32 + hi * 16);
        acc[G] = wmma16(afr[kc], bf, acc[G]);
      }
    }
    // LSTM cell (gate order i,f,g,o)
    float hn[8];
#pragma unroll
    for (int r = 0; r < 8; ++r) {
      float ig = sigmoidf_(acc[0][r]);
      float fg = sigmoidf_(acc[1][r]);
      float gg = tanhf(acc[2][r]);
      float og = sigmoidf_(acc[3][r]);
      float c = fg * creg[r] + ig * gg;
      creg[r] = c;
      hn[r] = og * tanhf(c);
    }
    // pooling score partials: per-row dot(h, pool_w)
#pragma unroll
    for (int r = 0; r < 8; ++r) {
      float v = rowred_sum(hn[r] * pw);
      if (m == 0) atomicAdd(&Slds[r + 8 * hi], v);
    }
    __syncthreads();
    if (tid < 16) {  // online softmax update per batch row
      float s = Slds[tid] + pb; Slds[tid] = 0.f;
      float mo = Mst[tid], mn = fmaxf(mo, s);
      float sc = __expf(mo - mn), wt = __expf(s - mn);
      Mst[tid] = mn; Sst[tid] = Sst[tid] * sc + wt;
      Scl[tid] = sc; Wgt[tid] = wt;
    }
    __syncthreads();
#pragma unroll
    for (int r = 0; r < 8; ++r) {
      int b = r + 8 * hi;
      float* ap = Acc + b * ACCS + dcol;
      *ap = (*ap) * Scl[b] + Wgt[b] * hn[r];
      Hlds[b * WS + dcol] = (_Float16)hn[r];
    }
    __syncthreads();
  }

  // C[b,n,:] = Acc / sum  -> workspace f16
  for (int idx = tid; idx < 16 * DDIM; idx += 256) {
    int b = idx >> 7, d = idx & 127;
    float v = Acc[b * ACCS + d] / Sst[b];
    Cf16[((size_t)(b0 + b) * NVARS + n) * DDIM + d] = (_Float16)v;
  }
}

// ---------------------------------------------------------------------------
// MHSA over the variable axis: one workgroup per batch row. qkv and out-proj
// as WMMA GEMMs, per-head 32x32 attention with WMMA (K = dh = 32).
// ---------------------------------------------------------------------------
__global__ void mhsa_kernel(const _Float16* __restrict__ Cf16,
                            const float* __restrict__ Wqkv,
                            const float* __restrict__ bqkv,
                            const float* __restrict__ Wo,
                            const float* __restrict__ bo,
                            float* __restrict__ CstarF,
                            _Float16* __restrict__ Cs16) {
  extern __shared__ char smem[];
  _Float16* Clds = (_Float16*)smem;        // 32*WS
  _Float16* Qlds = Clds + 32 * WS;         // 32*WS
  _Float16* Klds = Qlds + 32 * WS;         // 32*WS
  _Float16* Vt   = Klds + 32 * WS;         // 128*40  [dv][k] (V transposed)
  _Float16* Att  = Vt + 128 * 40;          // 8*16*40 per-wave attn staging
  _Float16* A2   = Att + 8 * 16 * 40;      // 32*WS   (attn output, pre out-proj)

  const int tid = threadIdx.x;
  const int wave = tid >> 5, lane = tid & 31, m = lane & 15, hi = lane >> 4;
  const int b = blockIdx.x;

  // load C rows for this batch element (f16, 16B vector copies)
  for (int idx = tid; idx < 32 * 16; idx += 256) {
    int row = idx >> 4, ch = idx & 15;
    const uint4* src = (const uint4*)(Cf16 + ((size_t)b * NVARS + row) * DDIM + ch * 8);
    *(uint4*)(Clds + row * WS + ch * 8) = *src;
  }
  __syncthreads();

  // qkv = C @ Wqkv^T + bqkv : 2 M-tiles x 24 col-tiles = 48 tiles
#pragma unroll
  for (int j = 0; j < 6; ++j) {
    int tt = wave + 8 * j;
    int mt = tt & 1, ot = tt >> 1;
    int o = ot * 16 + m;
    v8f acc;
    float bb = bqkv[o];
#pragma unroll
    for (int r = 0; r < 8; ++r) acc[r] = bb;
#pragma unroll
    for (int kc = 0; kc < 4; ++kc) {
      v16h af = load_a_frag(Clds + (mt * 16 + m) * WS, kc * 32, hi);
      v16h bf = load_b_frag_f32(Wqkv + (size_t)o * DDIM + kc * 32 + hi * 16);
      acc = wmma16(af, bf, acc);
    }
#pragma unroll
    for (int r = 0; r < 8; ++r) {
      int row = mt * 16 + r + 8 * hi;
      int col = ot * 16 + m;
      _Float16 hv = (_Float16)acc[r];
      if (ot < 8)       Qlds[row * WS + col] = hv;
      else if (ot < 16) Klds[row * WS + (col - 128)] = hv;
      else              Vt[(col - 256) * 40 + row] = hv;   // transpose V
    }
  }
  __syncthreads();

  // attention: wave -> (head = wave>>1, q-tile = wave&1)
  {
    int hd = wave >> 1, qt = wave & 1;
    int q0 = qt * 16, dh0 = hd * DH;
    v16h aq = load_a_frag(Qlds + (q0 + m) * WS, dh0, hi);
    v8f s0, s1;
#pragma unroll
    for (int r = 0; r < 8; ++r) { s0[r] = 0.f; s1[r] = 0.f; }
    {
      v16h bk0 = load_b_frag(Klds + m * WS + dh0 + hi * 16);
      v16h bk1 = load_b_frag(Klds + (16 + m) * WS + dh0 + hi * 16);
      s0 = wmma16(aq, bk0, s0);
      s1 = wmma16(aq, bk1, s1);
    }
    const float rs = 0.17677669529663687f;  // 1/sqrt(dh)
    _Float16* aw = Att + wave * 16 * 40;
#pragma unroll
    for (int r = 0; r < 8; ++r) {
      float a = s0[r] * rs, c = s1[r] * rs;
      float mx = rowred_max(fmaxf(a, c));
      float e0 = __expf(a - mx), e1 = __expf(c - mx);
      float inv = 1.f / rowred_sum(e0 + e1);
      int row = r + 8 * hi;
      aw[row * 40 + m] = (_Float16)(e0 * inv);
      aw[row * 40 + 16 + m] = (_Float16)(e1 * inv);
    }
    // A = attn @ V (K = 32 keys in one WMMA per 16-wide dv tile)
    v16h aa = load_a_frag(aw + m * 40, 0, hi);
#pragma unroll
    for (int jt = 0; jt < 2; ++jt) {
      int dv0 = dh0 + jt * 16;
      v16h bv = load_b_frag(Vt + (dv0 + m) * 40 + hi * 16);
      v8f o;
#pragma unroll
      for (int r = 0; r < 8; ++r) o[r] = 0.f;
      o = wmma16(aa, bv, o);
#pragma unroll
      for (int r = 0; r < 8; ++r)
        A2[(q0 + r + 8 * hi) * WS + dv0 + m] = (_Float16)o[r];
    }
  }
  __syncthreads();

  // out-proj: C_star = A @ Wo^T + bo : 2 M-tiles x 8 col-tiles
#pragma unroll
  for (int j = 0; j < 2; ++j) {
    int tt = wave + 8 * j;
    int mt = tt & 1, et = tt >> 1;
    int e = et * 16 + m;
    v8f acc;
    float bb = bo[e];
#pragma unroll
    for (int r = 0; r < 8; ++r) acc[r] = bb;
#pragma unroll
    for (int kc = 0; kc < 4; ++kc) {
      v16h af = load_a_frag(A2 + (mt * 16 + m) * WS, kc * 32, hi);
      v16h bf = load_b_frag_f32(Wo + (size_t)e * DDIM + kc * 32 + hi * 16);
      acc = wmma16(af, bf, acc);
    }
#pragma unroll
    for (int r = 0; r < 8; ++r) {
      int row = mt * 16 + r + 8 * hi;
      size_t off = ((size_t)b * NVARS + row) * DDIM + e;
      CstarF[off] = acc[r];
      Cs16[off] = (_Float16)acc[r];
    }
  }
}

// ---------------------------------------------------------------------------
// Decoder: per-(variable, batch-quarter) workgroup. WMMA init (tanh(C* @ W)),
// 128 LSTM steps with LDS-resident Whh, per-step output projection.
// ---------------------------------------------------------------------------
__global__ void dec_kernel(const _Float16* __restrict__ Cs16,
                           const float* __restrict__ ihW, const float* __restrict__ ihb,
                           const float* __restrict__ icW, const float* __restrict__ icb,
                           const float* __restrict__ Whh,
                           const float* __restrict__ bih, const float* __restrict__ bhh,
                           const float* __restrict__ outW, const float* __restrict__ outb,
                           float* __restrict__ out) {
  extern __shared__ char smem[];
  _Float16* Wlds = (_Float16*)smem;          // FOURD*WS
  _Float16* Hlds = Wlds + FOURD * WS;        // 16*WS
  float* Olds = (float*)(Hlds + 16 * WS);    // 16

  const int tid = threadIdx.x;
  const int wave = tid >> 5, lane = tid & 31, m = lane & 15, hi = lane >> 4;
  const int n = blockIdx.x >> 2, bq = blockIdx.x & 3;
  const int b0 = bq * 16;

  const float* Wg = Whh + (size_t)n * FOURD * DDIM;
  for (int idx = tid; idx < FOURD * DDIM / 4; idx += 256) {
    float4 f = ((const float4*)Wg)[idx];
    int e = idx * 4, g = e >> 7, d = e & 127;
    _Float16* p = Wlds + g * WS + d;
    p[0] = (_Float16)f.x; p[1] = (_Float16)f.y; p[2] = (_Float16)f.z; p[3] = (_Float16)f.w;
  }
  if (tid < 16) Olds[tid] = 0.f;

  const int dcol = wave * 16 + m;
  float bias_g[4];
#pragma unroll
  for (int G = 0; G < 4; ++G)
    bias_g[G] = bih[n * FOURD + G * DDIM + dcol] + bhh[n * FOURD + G * DDIM + dcol];
  const float ow = outW[n * DDIM + dcol];
  const float ob = outb[n];

  // h0 = tanh(C* @ ihW^T + ihb), c0 = tanh(C* @ icW^T + icb) via WMMA
  v8f ah, ac;
  {
    float bh_ = ihb[n * DDIM + dcol], bc_ = icb[n * DDIM + dcol];
#pragma unroll
    for (int r = 0; r < 8; ++r) { ah[r] = bh_; ac[r] = bc_; }
#pragma unroll
    for (int kc = 0; kc < 4; ++kc) {
      v16h af = load_a_frag(Cs16 + ((size_t)(b0 + m) * NVARS + n) * DDIM, kc * 32, hi);
      v16h bfh = load_b_frag_f32(ihW + ((size_t)n * DDIM + dcol) * DDIM + kc * 32 + hi * 16);
      v16h bfc = load_b_frag_f32(icW + ((size_t)n * DDIM + dcol) * DDIM + kc * 32 + hi * 16);
      ah = wmma16(af, bfh, ah);
      ac = wmma16(af, bfc, ac);
    }
  }
  float creg[8];
#pragma unroll
  for (int r = 0; r < 8; ++r) {
    creg[r] = tanhf(ac[r]);
    Hlds[(r + 8 * hi) * WS + dcol] = (_Float16)tanhf(ah[r]);
  }
  __syncthreads();

  for (int l = 0; l < LSEQ; ++l) {
    v16h afr[4];
#pragma unroll
    for (int kc = 0; kc < 4; ++kc) afr[kc] = load_a_frag(Hlds + m * WS, kc * 32, hi);

    v8f acc[4];
#pragma unroll
    for (int G = 0; G < 4; ++G) {
#pragma unroll
      for (int r = 0; r < 8; ++r) acc[G][r] = bias_g[G];
    }
    // kc outer / G inner: independent accumulators back-to-back (no WMMA nops)
#pragma unroll
    for (int kc = 0; kc < 4; ++kc) {
#pragma unroll
      for (int G = 0; G < 4; ++G) {
        const _Float16* brow = Wlds + (G * DDIM + dcol) * WS;
        v16h bf = load_b_frag(brow + kc * 32 + hi * 16);
        acc[G] = wmma16(afr[kc], bf, acc[G]);
      }
    }
    float hn[8];
#pragma unroll
    for (int r = 0; r < 8; ++r) {
      float ig = sigmoidf_(acc[0][r]);
      float fg = sigmoidf_(acc[1][r]);
      float gg = tanhf(acc[2][r]);
      float og = sigmoidf_(acc[3][r]);
      float c = fg * creg[r] + ig * gg;
      creg[r] = c;
      hn[r] = og * tanhf(c);
    }
    __syncthreads();  // all A-fragment reads of old h complete
#pragma unroll
    for (int r = 0; r < 8; ++r) {
      Hlds[(r + 8 * hi) * WS + dcol] = (_Float16)hn[r];
      float v = rowred_sum(hn[r] * ow);
      if (m == 0) atomicAdd(&Olds[r + 8 * hi], v);
    }
    __syncthreads();
    if (tid < 16) {
      float o = Olds[tid] + ob; Olds[tid] = 0.f;
      int bg = b0 + tid;
      if (l < LSEQ - 1)
        out[(size_t)bg * (LSEQ - 1) * NVARS + (size_t)l * NVARS + n] = o;      // recon
      else
        out[(size_t)BATCH * (LSEQ - 1) * NVARS + (size_t)bg * NVARS + n] = o;  // pred
    }
    __syncthreads();
  }
}

extern "C" void kernel_launch(void* const* d_in, const int* in_sizes, int n_in,
                              void* d_out, int out_size, void* d_ws, size_t ws_size,
                              hipStream_t stream) {
  (void)in_sizes; (void)n_in; (void)out_size; (void)ws_size;
  const float* X     = (const float*)d_in[0];
  const float* eWih  = (const float*)d_in[1];
  const float* eWhh  = (const float*)d_in[2];
  const float* ebih  = (const float*)d_in[3];
  const float* ebhh  = (const float*)d_in[4];
  const float* poolw = (const float*)d_in[5];
  const float* poolb = (const float*)d_in[6];
  const float* Wqkv  = (const float*)d_in[7];
  const float* bqkv  = (const float*)d_in[8];
  const float* Wo    = (const float*)d_in[9];
  const float* bo    = (const float*)d_in[10];
  const float* dihW  = (const float*)d_in[11];
  const float* dihb  = (const float*)d_in[12];
  const float* dicW  = (const float*)d_in[13];
  const float* dicb  = (const float*)d_in[14];
  const float* dWhh  = (const float*)d_in[16];
  const float* dbih  = (const float*)d_in[17];
  const float* dbhh  = (const float*)d_in[18];
  const float* doutW = (const float*)d_in[19];
  const float* doutb = (const float*)d_in[20];

  float* out = (float*)d_out;
  _Float16* Cf16 = (_Float16*)d_ws;                       // (B,N,D) f16
  _Float16* Cs16 = Cf16 + (size_t)BATCH * NVARS * DDIM;   // (B,N,D) f16
  float* CstarF = out + (size_t)BATCH * (LSEQ - 1) * NVARS + (size_t)BATCH * NVARS;

  size_t enc_smem  = (size_t)FOURD * WS * 2 + 16 * WS * 2 + 16 * ACCS * 4 + 6 * 16 * 4;
  size_t mhsa_smem = (size_t)32 * WS * 2 * 3 + 128 * 40 * 2 + 8 * 16 * 40 * 2 + 32 * WS * 2;
  size_t dec_smem  = (size_t)FOURD * WS * 2 + 16 * WS * 2 + 16 * 4;

  enc_kernel<<<dim3(NVARS * 4), dim3(256), enc_smem, stream>>>(
      X, eWih, eWhh, ebih, ebhh, poolw, poolb, Cf16);
  mhsa_kernel<<<dim3(BATCH), dim3(256), mhsa_smem, stream>>>(
      Cf16, Wqkv, bqkv, Wo, bo, CstarF, Cs16);
  dec_kernel<<<dim3(NVARS * 4), dim3(256), dec_smem, stream>>>(
      Cs16, dihW, dihb, dicW, dicb, dWhh, dbih, dbhh, doutW, doutb, out);
}
